// EntropyGuidance_76184129896856
// MI455X (gfx1250) — compile-verified, matrix operand order checked by reading, not run on hardware
//
#include <hip/hip_runtime.h>
#include <hip/hip_bf16.h>

typedef _Float16 v16h __attribute__((ext_vector_type(16)));
typedef _Float16 v8h  __attribute__((ext_vector_type(8)));
typedef _Float16 v4h  __attribute__((ext_vector_type(4)));
typedef float    v8f  __attribute__((ext_vector_type(8)));

#define B_    16
#define C_    64
#define HW_   16384
#define ROWS  (B_ * C_)          // 1024 rows per tensor
#define EPSF  1e-9f
#define PSCALE 4096.0f
#define INV_S2 (1.0f / (PSCALE * PSCALE))

// ---------------------------------------------------------------------------
// K0: per-row max and sum(exp(x-max)) for vis rows [0,1024) and text rows
// [1024,2048). One 256-thread block per row.
// ---------------------------------------------------------------------------
__global__ void k_rowstats(const float* __restrict__ vis,
                           const float* __restrict__ text,
                           float* __restrict__ rmax, float* __restrict__ rZ) {
    __shared__ float red[256];
    const int row = blockIdx.x;          // 0..2047
    const int tid = threadIdx.x;
    const float* src = (row < ROWS ? vis : text) + (size_t)(row & (ROWS - 1)) * HW_;
    const float4* s4 = (const float4*)src;

    float m = -3.4e38f;
    for (int i = tid; i < HW_ / 4; i += 256) {
        float4 x = s4[i];
        m = fmaxf(m, fmaxf(fmaxf(x.x, x.y), fmaxf(x.z, x.w)));
    }
    red[tid] = m; __syncthreads();
    for (int s = 128; s > 0; s >>= 1) {
        if (tid < s) red[tid] = fmaxf(red[tid], red[tid + s]);
        __syncthreads();
    }
    m = red[0]; __syncthreads();

    float z = 0.f;
    for (int i = tid; i < HW_ / 4; i += 256) {
        float4 x = s4[i];
        z += expf(x.x - m) + expf(x.y - m) + expf(x.z - m) + expf(x.w - m);
    }
    red[tid] = z; __syncthreads();
    for (int s = 128; s > 0; s >>= 1) {
        if (tid < s) red[tid] += red[tid + s];
        __syncthreads();
    }
    if (tid == 0) { rmax[row] = m; rZ[row] = red[0]; }
}

// ---------------------------------------------------------------------------
// K1: write softmax probabilities as f16 (scaled by PSCALE), accumulate row
// mean of p and (for text rows) the spatial entropy. Also zero mi[].
// ---------------------------------------------------------------------------
__global__ void k_softmax(const float* __restrict__ vis,
                          const float* __restrict__ text,
                          _Float16* __restrict__ pv16, _Float16* __restrict__ pt16,
                          const float* __restrict__ rmax, const float* __restrict__ rZ,
                          float* __restrict__ rmean, float* __restrict__ ent,
                          float* __restrict__ mi) {
    __shared__ float redp[256];
    __shared__ float rede[256];
    const int row = blockIdx.x;          // 0..2047
    const int tid = threadIdx.x;
    const bool isText = row >= ROWS;
    const int r = row & (ROWS - 1);
    const float4* s4 = (const float4*)((isText ? text : vis) + (size_t)r * HW_);
    v4h* dst = (v4h*)((isText ? pt16 : pv16) + (size_t)r * HW_);

    const float m = rmax[row];
    const float inv = 1.0f / rZ[row];
    float ps = 0.f, es = 0.f;
    for (int i = tid; i < HW_ / 4; i += 256) {
        float4 x = s4[i];
        float p0 = expf(x.x - m) * inv;
        float p1 = expf(x.y - m) * inv;
        float p2 = expf(x.z - m) * inv;
        float p3 = expf(x.w - m) * inv;
        ps += (p0 + p1) + (p2 + p3);
        if (isText) {
            es -= p0 * logf(p0 + EPSF) + p1 * logf(p1 + EPSF)
                + p2 * logf(p2 + EPSF) + p3 * logf(p3 + EPSF);
        }
        v4h o;
        o[0] = (_Float16)(p0 * PSCALE);
        o[1] = (_Float16)(p1 * PSCALE);
        o[2] = (_Float16)(p2 * PSCALE);
        o[3] = (_Float16)(p3 * PSCALE);
        dst[i] = o;
    }
    redp[tid] = ps; rede[tid] = es; __syncthreads();
    for (int s = 128; s > 0; s >>= 1) {
        if (tid < s) { redp[tid] += redp[tid + s]; rede[tid] += rede[tid + s]; }
        __syncthreads();
    }
    if (tid == 0) {
        rmean[row] = redp[0] * (1.0f / HW_);
        if (isText) ent[r] = rede[0];
    }
    if (blockIdx.x == 0 && tid < B_) mi[tid] = 0.f;
}

// ---------------------------------------------------------------------------
// K2: joint[b] = Pvis @ Ptext^T via v_wmma_f32_16x16x32_f16, then the MI
// contribution of this 16x16 tile -> atomicAdd into mi[b].
// Grid: (16 tiles, 16 batches), 128 threads (4 waves, K split 4 ways).
// Fragment layout per ISA 7.12.2 (16-bit A 16x32): lane r (0..15) holds row
// M=r; lanes 0-15 carry K = kb+0..7 and kb+16..23, lanes 16-31 carry
// K = kb+8..15 and kb+24..31. B mirrored with N in place of M.
// ---------------------------------------------------------------------------
__global__ void k_mi(const _Float16* __restrict__ pv16,
                     const _Float16* __restrict__ pt16,
                     const float* __restrict__ rmean,
                     float* __restrict__ mi) {
    __shared__ float part[4 * 256];
    __shared__ float red[128];
    const int tid  = threadIdx.x;
    const int w    = tid >> 5;
    const int lane = tid & 31;
    const int half = lane >> 4;
    const int r    = lane & 15;
    const int tile = blockIdx.x;         // 0..15
    const int b    = blockIdx.y;         // 0..15
    const int mt   = tile >> 2;
    const int nt   = tile & 3;

    const _Float16* Ap = pv16 + (size_t)(b * C_ + mt * 16 + r) * HW_;
    const _Float16* Bp = pt16 + (size_t)(b * C_ + nt * 16 + r) * HW_;

    v8f acc = {};
    const int k0 = w * (HW_ / 4);
    for (int kb = k0; kb < k0 + HW_ / 4; kb += 32) {
        v8h alo = *(const v8h*)(Ap + kb + half * 8);
        v8h ahi = *(const v8h*)(Ap + kb + 16 + half * 8);
        v8h blo = *(const v8h*)(Bp + kb + half * 8);
        v8h bhi = *(const v8h*)(Bp + kb + 16 + half * 8);
        v16h a = __builtin_shufflevector(alo, ahi, 0,1,2,3,4,5,6,7,8,9,10,11,12,13,14,15);
        v16h bm = __builtin_shufflevector(blo, bhi, 0,1,2,3,4,5,6,7,8,9,10,11,12,13,14,15);
        acc = __builtin_amdgcn_wmma_f32_16x16x32_f16(
            /*neg_a=*/false, a, /*neg_b=*/false, bm,
            /*c_mod=*/(short)0, acc, /*reuse_a=*/false, /*reuse_b=*/false);
    }
#pragma unroll
    for (int i = 0; i < 8; ++i)
        part[w * 256 + (i + 8 * half) * 16 + r] = acc[i];
    __syncthreads();

    float tsum = 0.f;
    for (int e = tid; e < 256; e += 128) {
        float j = part[e] + part[256 + e] + part[512 + e] + part[768 + e];
        j *= INV_S2;                                  // undo f16 scaling
        const int M = e >> 4, N = e & 15;
        const float pvm = rmean[b * C_ + mt * 16 + M];
        const float ptm = rmean[ROWS + b * C_ + nt * 16 + N];
        tsum += j * logf(j / (pvm * ptm) + EPSF);
    }
    red[tid] = tsum; __syncthreads();
    for (int s = 64; s > 0; s >>= 1) {
        if (tid < s) red[tid] += red[tid + s];
        __syncthreads();
    }
    if (tid == 0) atomicAdd(mi + b, red[0]);
}

// ---------------------------------------------------------------------------
// K3: out = vis + sigmoid(1 - ent[b,c] + 0.5*mi[b]) * text, float4-vectorized.
// ---------------------------------------------------------------------------
__global__ void k_final(const float* __restrict__ vis,
                        const float* __restrict__ text,
                        const float* __restrict__ ent,
                        const float* __restrict__ mi,
                        float* __restrict__ out) {
    const int idx = blockIdx.x * 256 + threadIdx.x;   // float4 index
    const int bc = idx >> 12;                         // 4096 float4 per (b,c) row
    const int b = bc >> 6;
    const float z = 1.0f - ent[bc] + 0.5f * mi[b];
    const float g = 1.0f / (1.0f + expf(-z));
    float4 v = ((const float4*)vis)[idx];
    float4 t = ((const float4*)text)[idx];
    float4 o;
    o.x = v.x + g * t.x;
    o.y = v.y + g * t.y;
    o.z = v.z + g * t.z;
    o.w = v.w + g * t.w;
    ((float4*)out)[idx] = o;
}

extern "C" void kernel_launch(void* const* d_in, const int* in_sizes, int n_in,
                              void* d_out, int out_size, void* d_ws, size_t ws_size,
                              hipStream_t stream) {
    const float* vis  = (const float*)d_in[0];
    const float* text = (const float*)d_in[1];
    float* out = (float*)d_out;

    // Workspace layout: pv16 (32MB) | pt16 (32MB) | rmax[2048] rZ[2048]
    //                   rmean[2048] ent[1024] mi[16]
    _Float16* pv16 = (_Float16*)d_ws;
    _Float16* pt16 = pv16 + (size_t)ROWS * HW_;
    float* stats = (float*)(pt16 + (size_t)ROWS * HW_);
    float* rmax  = stats;
    float* rZ    = stats + 2048;
    float* rmean = stats + 4096;
    float* ent   = stats + 6144;
    float* mi    = stats + 7168;

    k_rowstats<<<2 * ROWS, 256, 0, stream>>>(vis, text, rmax, rZ);
    k_softmax <<<2 * ROWS, 256, 0, stream>>>(vis, text, pv16, pt16, rmax, rZ,
                                             rmean, ent, mi);
    k_mi      <<<dim3(16, 16), 128, 0, stream>>>(pv16, pt16, rmean, mi);
    k_final   <<<(B_ * C_ * HW_ / 4) / 256, 256, 0, stream>>>(vis, text, ent, mi, out);
}
